// MPNNGNN_1881195675865
// MI455X (gfx1250) — matrix operand order repaired and checked by reading, not compiled, not run on hardware
//
#include <hip/hip_runtime.h>
#include <hip/hip_bf16.h>

#define NN 20000
#define EE 100000
#define DD 64
#define HH 128
#define NODE_IN 74
#define EDGE_IN 12
#define NUM_STEPS 6

// msg GEMM: K_total = 128*64 (u (x) h outer product) + 64 (be2 rows, u==1)
#define CH_MSG 258   // K chunks of 32
#define FT_MSG 4     // 64 output cols / 16
#define TPW 5        // edge tiles per wave (6250 = 1250 * 5)
// GRU GEMMs: K_total = 64, N = 192
#define CH_G 2
#define FT_G 12

typedef float v8f __attribute__((ext_vector_type(8)));
typedef __bf16 v16bf __attribute__((ext_vector_type(16)));
typedef unsigned int v8u __attribute__((ext_vector_type(8)));

__device__ __forceinline__ unsigned short f32_to_bf16(float f) {
  unsigned int u = __float_as_uint(f);
  u += 0x7fffu + ((u >> 16) & 1u);           // round-to-nearest-even
  return (unsigned short)(u >> 16);
}
__device__ __forceinline__ float bf16_to_f32(unsigned short s) {
  return __uint_as_float(((unsigned int)s) << 16);
}
__device__ __forceinline__ v8f splat8(float x) {
  v8f r;
#pragma unroll
  for (int i = 0; i < 8; ++i) r[i] = x;
  return r;
}

// Build a 16x32 bf16 A-fragment (wave32 WMMA) from one bf16 row in LDS.
// rowbase points at this lane's row M = lane%16. Elements scaled by `scale`.
// 16-bit A 16x32 (contiguous K order, matching B swizzle): lane<16: halves 0..15 -> K=0..15;
// lane>=16: halves 0..15 -> K=16..31.
// Packing: two scaled f32 products -> one dword via v_perm_b32 (bytes 3:2 of
// each), with +0x8000 round-half bias. ~7 VALU per packed pair.
__device__ __forceinline__ v16bf build_a_frag(const unsigned short* rowbase,
                                              int kbase, int lane, float scale) {
  int off = kbase + ((lane & 16) ? 16 : 0);
  uint4 p0 = *(const uint4*)(rowbase + off);        // 8 bf16, ds_load_b128
  uint4 p1 = *(const uint4*)(rowbase + off + 8);    // 8 bf16
  unsigned int w[8] = {p0.x, p0.y, p0.z, p0.w, p1.x, p1.y, p1.z, p1.w};
  v8u a;
#pragma unroll
  for (int j = 0; j < 8; ++j) {
    float lo = __uint_as_float(w[j] << 16) * scale;
    float hi = __uint_as_float(w[j] & 0xffff0000u) * scale;
    unsigned int lb = __float_as_uint(lo) + 0x8000u;
    unsigned int hb = __float_as_uint(hi) + 0x8000u;
    a[j] = __builtin_amdgcn_perm(hb, lb, 0x07060302u);  // {hb.hi16, lb.hi16}
  }
  return __builtin_bit_cast(v16bf, a);
}

// Load one pre-swizzled 32x16 bf16 B-fragment: contiguous 32B per lane.
__device__ __forceinline__ v16bf load_b_frag(const unsigned short* Bsw,
                                             int fragIdx, int lane) {
  const v8u* p = (const v8u*)(Bsw + ((size_t)fragIdx * 32 + lane) * 16);
  return __builtin_bit_cast(v16bf, *p);
}

__device__ __forceinline__ v8f wmma_bf16(v16bf a, v16bf b, v8f c) {
  return __builtin_amdgcn_wmma_f32_16x16x32_bf16(false, a, false, b,
                                                 (short)0, c, false, false);
}

// ---------------------------------------------------------------- utility
__global__ void k_zero(float* p, int n) {
  int i = blockIdx.x * 256 + threadIdx.x;
  if (i < n) p[i] = 0.0f;
}

// h = relu(node_feats @ W_proj + b_proj)   (one-time, 189 MFLOP -> VALU f32)
__global__ void k_node_proj(const float* nf, const float* W, const float* b,
                            float* h) {
  int i = blockIdx.x * 256 + threadIdx.x;   // < NN*DD
  int n = i >> 6, d = i & 63;
  const float* row = nf + (size_t)n * NODE_IN;
  float acc = b[d];
#pragma unroll 2
  for (int k = 0; k < NODE_IN; ++k) acc = fmaf(row[k], W[k * DD + d], acc);
  h[i] = acc > 0.0f ? acc : 0.0f;
}

// u = relu(edge_feats @ We1 + be1) stored bf16 (one-time, 0.3 GFLOP)
__global__ void k_edge_mlp1(const float* ef, const float* W, const float* b,
                            unsigned short* u) {
  int i = blockIdx.x * 256 + threadIdx.x;   // < EE*HH
  int e = i >> 7, j = i & 127;
  const float* row = ef + (size_t)e * EDGE_IN;
  float acc = b[j];
#pragma unroll
  for (int k = 0; k < EDGE_IN; ++k) acc = fmaf(row[k], W[k * HH + j], acc);
  u[i] = f32_to_bf16(acc > 0.0f ? acc : 0.0f);
}

// Pre-swizzle We2 (viewed [8192,64]) + be2 (viewed [64,64]) into WMMA
// B-fragment order. Frag element (c,t,lane,hi) <- B[c*32 + K, t*16 + N],
// K = (lane<16) ? hi : 16+hi, N = lane%16.
__global__ void k_preswz_msg(const float* We2, const float* be2,
                             unsigned short* dstp) {
  int idx = blockIdx.x * 256 + threadIdx.x;   // < CH_MSG*FT_MSG*512
  int hi = idx & 15, l = (idx >> 4) & 31, rest = idx >> 9;
  int t = rest & 3, c = rest >> 2;
  int Krow = c * 32 + ((l < 16) ? hi : 16 + hi);
  int f = t * 16 + (l & 15);
  float v = (Krow < HH * DD) ? We2[(size_t)Krow * DD + f]
                             : be2[(size_t)(Krow - HH * DD) * DD + f];
  dstp[idx] = f32_to_bf16(v);
}

// Pre-swizzle Wih/Whh ([192,64] torch layout) transposed: B[d,j] = W[j*64+d]
__global__ void k_preswz_gru(const float* W, unsigned short* dstp) {
  int idx = blockIdx.x * 256 + threadIdx.x;   // < CH_G*FT_G*512
  int hi = idx & 15, l = (idx >> 4) & 31, rest = idx >> 9;
  int t = rest % FT_G, c = rest / FT_G;
  int Krow = c * 32 + ((l < 16) ? hi : 16 + hi);  // d, < 64
  int col = t * 16 + (l & 15);                    // j, < 192
  dstp[idx] = f32_to_bf16(W[(size_t)col * DD + Krow]);
}

// ------------------------------------------------------- per-step kernels
// msg[e,:] = (u[e] (x) h[src[e]]) @ We2view + h[src[e]] @ be2view,
// scatter-added to agg[dst[e]]. One wave per TPW 16-edge tiles: B fragments
// are register double-buffered and reused across TPW A tiles (5x less L2
// B traffic, loads for chunk c+1 in flight during chunk c's 20 WMMAs).
__global__ void __launch_bounds__(32)
k_msg(const float* __restrict__ h, const unsigned short* __restrict__ u_bf,
      const int* __restrict__ src, const int* __restrict__ dstI,
      const unsigned short* __restrict__ Bsw, float* __restrict__ agg) {
  __shared__ __align__(16) unsigned short lds_u[TPW * 16 * HH];
  __shared__ __align__(16) unsigned short lds_h[TPW * 16 * DD];
  int lane = threadIdx.x;
  int ebase = blockIdx.x * (16 * TPW);

  // Stage u rows (contiguous TPW*16*128 bf16)
  const unsigned int* gu = (const unsigned int*)(u_bf + (size_t)ebase * HH);
  unsigned int* lu = (unsigned int*)lds_u;
#pragma unroll
  for (int i = 0; i < TPW * 32; ++i) lu[lane + 32 * i] = gu[lane + 32 * i];
  // Stage gathered h[src] rows as bf16
#pragma unroll
  for (int i = 0; i < TPW * 32; ++i) {
    int t = lane + 32 * i;            // 0..TPW*1024-1
    int e = t >> 6, d = t & 63;
    int srow = src[ebase + e];
    lds_h[t] = f32_to_bf16(h[(size_t)srow * DD + d]);
  }
  asm volatile("s_wait_dscnt 0" ::: "memory");  // LDS staging visible wave-wide

  const unsigned short* hrow = lds_h + (lane & 15) * DD;
  const unsigned short* urow = lds_u + (lane & 15) * HH;

  v8f acc[TPW][FT_MSG];
#pragma unroll
  for (int t = 0; t < TPW; ++t)
#pragma unroll
    for (int n = 0; n < FT_MSG; ++n) acc[t][n] = splat8(0.0f);

  // B double buffer: preload chunk 0
  v16bf b0 = load_b_frag(Bsw, 0, lane);
  v16bf b1 = load_b_frag(Bsw, 1, lane);
  v16bf b2 = load_b_frag(Bsw, 2, lane);
  v16bf b3 = load_b_frag(Bsw, 3, lane);

  for (int c = 0; c < CH_MSG; ++c) {
    // issue next chunk's B loads before consuming current (clamped at end)
    int cn = (c + 1 < CH_MSG) ? c + 1 : c;
    v16bf n0 = load_b_frag(Bsw, cn * FT_MSG + 0, lane);
    v16bf n1 = load_b_frag(Bsw, cn * FT_MSG + 1, lane);
    v16bf n2 = load_b_frag(Bsw, cn * FT_MSG + 2, lane);
    v16bf n3 = load_b_frag(Bsw, cn * FT_MSG + 3, lane);

    float us; int kb;
    if (c < 256) { kb = (c & 1) * 32; }
    else         { kb = (c - 256) * 32; }
#pragma unroll
    for (int t = 0; t < TPW; ++t) {
      if (c < 256) us = bf16_to_f32(urow[t * (16 * HH) + (c >> 1)]);
      else         us = 1.0f;
      v16bf a = build_a_frag(hrow + t * (16 * DD), kb, lane, us);
      acc[t][0] = wmma_bf16(a, b0, acc[t][0]);
      acc[t][1] = wmma_bf16(a, b1, acc[t][1]);
      acc[t][2] = wmma_bf16(a, b2, acc[t][2]);
      acc[t][3] = wmma_bf16(a, b3, acc[t][3]);
    }
    b0 = n0; b1 = n1; b2 = n2; b3 = n3;
  }

  // Scatter: C/D layout: lane<16 -> M=v, N=lane; lane>=16 -> M=8+v, N=lane-16
  int Nc = lane & 15;
  int Mo = (lane & 16) ? 8 : 0;
#pragma unroll
  for (int t = 0; t < TPW; ++t) {
#pragma unroll
    for (int v = 0; v < 8; ++v) {
      int M = v + Mo;
      int drow = dstI[ebase + t * 16 + M];
      float* base = agg + (size_t)drow * DD + Nc;
#pragma unroll
      for (int n = 0; n < FT_MSG; ++n) atomicAdd(base + n * 16, acc[t][n][v]);
    }
  }
}

// GRU cell for a 16-node tile: x = relu(agg + b_conv) (agg re-zeroed),
// gi = x@Wih^T + bih, gh = h@Whh^T + bhh, torch gate order r,z,n.
// All 16 accumulators live across the K loop; A fragments built once per
// K chunk; 48 WMMAs per tile issue back-to-back.
__global__ void __launch_bounds__(32)
k_gru(float* __restrict__ h, float* __restrict__ agg,
      const float* __restrict__ b_conv,
      const unsigned short* __restrict__ Bih,
      const unsigned short* __restrict__ Bhh,
      const float* __restrict__ bih, const float* __restrict__ bhh) {
  __shared__ __align__(16) unsigned short lds_x[16 * DD];
  __shared__ __align__(16) unsigned short lds_hb[16 * DD];
  int lane = threadIdx.x;
  int nbase = blockIdx.x * 16;

#pragma unroll
  for (int i = 0; i < 32; ++i) {
    int t = lane + 32 * i;            // 0..1023
    int nl = t >> 6, d = t & 63;
    size_t gidx = (size_t)(nbase + nl) * DD + d;
    float xv = agg[gidx] + b_conv[d];
    agg[gidx] = 0.0f;                 // reset accumulator for next step
    lds_x[t] = f32_to_bf16(xv > 0.0f ? xv : 0.0f);
    lds_hb[t] = f32_to_bf16(h[gidx]);
  }
  asm volatile("s_wait_dscnt 0" ::: "memory");

  const unsigned short* xrow = lds_x + (lane & 15) * DD;
  const unsigned short* hrow = lds_hb + (lane & 15) * DD;
  int Nc = lane & 15;
  int Mo = (lane & 16) ? 8 : 0;

  v8f ar[4], az[4], ai[4], an[4];
#pragma unroll
  for (int j0 = 0; j0 < 4; ++j0) {
    int f = j0 * 16 + Nc;
    ar[j0] = splat8(bih[f] + bhh[f]);
    az[j0] = splat8(bih[DD + f] + bhh[DD + f]);
    ai[j0] = splat8(bih[2 * DD + f]);
    an[j0] = splat8(bhh[2 * DD + f]);
  }
#pragma unroll
  for (int c = 0; c < CH_G; ++c) {
    v16bf ax = build_a_frag(xrow, c * 32, lane, 1.0f);
    v16bf ag = build_a_frag(hrow, c * 32, lane, 1.0f);
#pragma unroll
    for (int j0 = 0; j0 < 4; ++j0) {
      ar[j0] = wmma_bf16(ax, load_b_frag(Bih, c * FT_G + j0, lane), ar[j0]);
      ar[j0] = wmma_bf16(ag, load_b_frag(Bhh, c * FT_G + j0, lane), ar[j0]);
      az[j0] = wmma_bf16(ax, load_b_frag(Bih, c * FT_G + j0 + 4, lane), az[j0]);
      az[j0] = wmma_bf16(ag, load_b_frag(Bhh, c * FT_G + j0 + 4, lane), az[j0]);
      ai[j0] = wmma_bf16(ax, load_b_frag(Bih, c * FT_G + j0 + 8, lane), ai[j0]);
      an[j0] = wmma_bf16(ag, load_b_frag(Bhh, c * FT_G + j0 + 8, lane), an[j0]);
    }
  }
#pragma unroll
  for (int j0 = 0; j0 < 4; ++j0) {
    int f = j0 * 16 + Nc;
#pragma unroll
    for (int v = 0; v < 8; ++v) {
      int M = v + Mo;
      size_t gidx = (size_t)(nbase + M) * DD + f;
      float r = 1.0f / (1.0f + __expf(-ar[j0][v]));
      float z = 1.0f / (1.0f + __expf(-az[j0][v]));
      float n = tanhf(ai[j0][v] + r * an[j0][v]);
      float ho = h[gidx];
      h[gidx] = (1.0f - z) * n + z * ho;
    }
  }
}

// ------------------------------------------------------------------ host
extern "C" void kernel_launch(void* const* d_in, const int* in_sizes, int n_in,
                              void* d_out, int out_size, void* d_ws,
                              size_t ws_size, hipStream_t stream) {
  const float* node_feats = (const float*)d_in[0];
  const float* edge_feats = (const float*)d_in[1];
  const int*   src        = (const int*)d_in[2];
  const int*   dstI       = (const int*)d_in[3];
  const float* W_proj     = (const float*)d_in[4];
  const float* b_proj     = (const float*)d_in[5];
  const float* We1        = (const float*)d_in[6];
  const float* be1        = (const float*)d_in[7];
  const float* We2        = (const float*)d_in[8];
  const float* be2        = (const float*)d_in[9];
  const float* b_conv     = (const float*)d_in[10];
  const float* Wih        = (const float*)d_in[11];
  const float* Whh        = (const float*)d_in[12];
  const float* bih        = (const float*)d_in[13];
  const float* bhh        = (const float*)d_in[14];
  (void)in_sizes; (void)n_in; (void)out_size; (void)ws_size;

  char* p = (char*)d_ws;
  float* h   = (float*)p;                 p += (size_t)NN * DD * 4;
  float* agg = (float*)p;                 p += (size_t)NN * DD * 4;
  unsigned short* u_bf    = (unsigned short*)p; p += (size_t)EE * HH * 2;
  unsigned short* Bsw_msg = (unsigned short*)p; p += (size_t)CH_MSG * FT_MSG * 512 * 2;
  unsigned short* Bsw_ih  = (unsigned short*)p; p += (size_t)CH_G * FT_G * 512 * 2;
  unsigned short* Bsw_hh  = (unsigned short*)p; p += (size_t)CH_G * FT_G * 512 * 2;
  // total ~35 MB of workspace (fits easily in the 192 MB L2)

  k_zero<<<(NN * DD + 255) / 256, 256, 0, stream>>>(agg, NN * DD);
  k_node_proj<<<(NN * DD) / 256, 256, 0, stream>>>(node_feats, W_proj, b_proj, h);
  k_edge_mlp1<<<(EE * HH) / 256, 256, 0, stream>>>(edge_feats, We1, be1, u_bf);
  k_preswz_msg<<<(CH_MSG * FT_MSG * 512) / 256, 256, 0, stream>>>(We2, be2, Bsw_msg);
  k_preswz_gru<<<(CH_G * FT_G * 512) / 256, 256, 0, stream>>>(Wih, Bsw_ih);
  k_preswz_gru<<<(CH_G * FT_G * 512) / 256, 256, 0, stream>>>(Whh, Bsw_hh);

  for (int s = 0; s < NUM_STEPS; ++s) {
    k_msg<<<EE / (16 * TPW), 32, 0, stream>>>(h, u_bf, src, dstI, Bsw_msg, agg);
    k_gru<<<NN / 16, 32, 0, stream>>>(h, agg, b_conv, Bsw_ih, Bsw_hh, bih, bhh);
  }

  hipMemcpyAsync(d_out, h, (size_t)NN * DD * 4, hipMemcpyDeviceToDevice, stream);
}
